// SingleAttention_25726854103499
// MI455X (gfx1250) — matrix operand order repaired and checked by reading, not compiled, not run on hardware
//
#include <hip/hip_runtime.h>
#include <hip/hip_bf16.h>
#include <math.h>

#define B_  8
#define S_  2048
#define D_  256
#define DK_ 128
#define DV_ 128

typedef __attribute__((ext_vector_type(16))) _Float16     v16h;
typedef __attribute__((ext_vector_type(8)))  _Float16     v8h;
typedef __attribute__((ext_vector_type(8)))  float        v8f;
typedef __attribute__((ext_vector_type(4)))  unsigned int v4u;
typedef __attribute__((ext_vector_type(4)))  int          v4i;
typedef __attribute__((ext_vector_type(8)))  int          v8i;

union AFrag { v16h v; v8h h[2]; };

__device__ inline v8f vzero8f() {
    v8f z;
#pragma unroll
    for (int i = 0; i < 8; ++i) z[i] = 0.0f;
    return z;
}

// ---------------------------------------------------------------------------
// TDM: issue a 2D tensor-tile DMA (global f16 -> LDS), with optional LDS row
// padding for bank-conflict avoidance.  D# packed per CDNA5 ISA §8 (group0:
// count / lds_addr / global_addr / type=2; group1: data_size=2B, pad fields,
// tensor dims, tile dims, dim0 stride).  All args wave-uniform -> SGPRs.
// ---------------------------------------------------------------------------
__device__ inline void tdm_load_2d_f16(unsigned int lds_off,
                                       const _Float16* gptr,
                                       unsigned int tensor_d0, unsigned int tensor_d1,
                                       unsigned int tile_d0,   unsigned int tile_d1,
                                       unsigned int stride0,
                                       unsigned int pad_interval_code,
                                       unsigned int pad_amount_code,
                                       unsigned int pad_enable) {
    unsigned long long ga = (unsigned long long)(uintptr_t)gptr;
    v4u g0;
    g0[0] = 1u;                                                   // count=1
    g0[1] = lds_off;                                              // lds_addr
    g0[2] = (unsigned int)ga;                                     // global[31:0]
    g0[3] = (unsigned int)((ga >> 32) & 0x01FFFFFFu) | (2u << 30);// global[56:32]+type=2
    v8i g1;
    g1[0] = (int)((1u << 16) | (pad_enable << 20) |
                  (pad_interval_code << 22) | (pad_amount_code << 25)); // data_size=1 (2B)
    g1[1] = (int)((tensor_d0 & 0xFFFFu) << 16);                   // tensor_dim0[15:0]
    g1[2] = (int)(((tensor_d0 >> 16) & 0xFFFFu) |
                  ((tensor_d1 & 0xFFFFu) << 16));
    g1[3] = (int)(((tensor_d1 >> 16) & 0xFFFFu) |
                  ((tile_d0 & 0xFFFFu) << 16));                   // tile_dim0
    g1[4] = (int)(tile_d1 & 0xFFFFu);                             // tile_dim1, tile_dim2=0
    g1[5] = (int)stride0;                                         // dim0_stride[31:0]
    g1[6] = 0;                                                    // stride hi / dim1_stride lo
    g1[7] = 0;
    v4i g2 = {0, 0, 0, 0};
    v4i g3 = {0, 0, 0, 0};
#if __clang_major__ >= 23
    v8i g4 = {0, 0, 0, 0, 0, 0, 0, 0};
    __builtin_amdgcn_tensor_load_to_lds(g0, g1, g2, g3, g4, 0);
#else
    __builtin_amdgcn_tensor_load_to_lds(g0, g1, g2, g3, 0);
#endif
}

__device__ inline unsigned int lds_off_of(const void* p) {
    // Generic LDS pointer low 32 bits == LDS byte offset (ISA aperture rule).
    return (unsigned int)(uintptr_t)p;
}

// ---------------------------------------------------------------------------
// Kernel 0: transpose + convert a [D, DK] fp32 weight into [DK, D] f16 so the
// projection GEMM's B-fragments are contiguous 32B loads per lane.
// ---------------------------------------------------------------------------
__global__ void wt_convert_kernel(const float* __restrict__ W,
                                  _Float16* __restrict__ Wt) {
    int idx = blockIdx.x * blockDim.x + threadIdx.x;   // over D_*DK_
    if (idx >= D_ * DK_) return;
    int n = idx % DK_;
    int k = idx / DK_;
    Wt[(size_t)n * D_ + k] = (_Float16)W[(size_t)k * DK_ + n];
}

// ---------------------------------------------------------------------------
// Kernel 1: projection GEMM  out = X[B*S, D] @ W[D, DK] + bias  (WMMA f16)
// Each wave computes a 16-row x 128-col tile; 8 waves / 256-thread block.
// transposeOut==1 writes out as [B, DK, S] (for V, so P*V B-frags are
// contiguous); else [B*S, DK] row-major f16.
// ---------------------------------------------------------------------------
__global__ __launch_bounds__(256)
void proj16_kernel(const float* __restrict__ X,
                   const _Float16* __restrict__ Wt,
                   const float* __restrict__ bias,
                   _Float16* __restrict__ Out,
                   int transposeOut) {
    const int lane = threadIdx.x & 31;
    const int wave = threadIdx.x >> 5;
    const int gw   = blockIdx.x * 8 + wave;
    const int m0   = gw * 16;
    const int half = lane >> 4;
    const int ln   = lane & 15;
    const int row  = m0 + ln;                 // A-matrix row held by this lane

    v8f acc[8];
#pragma unroll
    for (int i = 0; i < 8; ++i) acc[i] = vzero8f();

    for (int kc = 0; kc < 8; ++kc) {          // K = 256 in chunks of 32
        const int k0 = kc * 32;
        // A-fragment: halves 0..7 <- K = k0+half*16.., halves 8..15 <- K = k0+half*16+8..
        const float* s0 = X + (size_t)row * D_ + k0 + half * 16;
        const float* s1 = s0 + 8;
        AFrag a;
#pragma unroll
        for (int i = 0; i < 8; ++i) {
            a.v[i]     = (_Float16)s0[i];
            a.v[8 + i] = (_Float16)s1[i];
        }
#pragma unroll
        for (int nt = 0; nt < 8; ++nt) {      // 8 N-tiles of 16 cols
            const _Float16* bp = Wt + (size_t)(nt * 16 + ln) * D_ + k0 + half * 16;
            v16h bfrag = *(const v16h*)bp;    // contiguous 32B per lane
            acc[nt] = __builtin_amdgcn_wmma_f32_16x16x32_f16(
                false, a.v, false, bfrag, (short)0, acc[nt], false, false);
        }
    }

    // Epilogue: C layout -> VGPR j holds row (j + half*8), col = nt*16 + ln
#pragma unroll
    for (int j = 0; j < 8; ++j) {
        const int m    = j + half * 8;
        const int grow = m0 + m;
#pragma unroll
        for (int nt = 0; nt < 8; ++nt) {
            const int n = nt * 16 + ln;
            float val = acc[nt][j] + bias[n];
            if (!transposeOut) {
                Out[(size_t)grow * DK_ + n] = (_Float16)val;
            } else {
                int bb = grow / S_, s = grow % S_;
                Out[((size_t)bb * DK_ + n) * S_ + s] = (_Float16)val;
            }
        }
    }
}

// ---------------------------------------------------------------------------
// Kernel 2: flash attention with TDM-staged K/V tiles.
// Block = 4 waves = 64 query rows (single batch).  Per 32-key iteration:
//   wave0 issues 2 TDM loads (next K tile 32x128, next V tile 128x32) into the
//   alternate LDS buffers (double-buffered, row-padded 16B by TDM pad_enable
//   for bank-conflict-free B-fragment reads), all waves compute 8 WMMA scores
//   + online softmax (shfl_xor row reductions) + LDS P re-layout + 8 WMMA P*V,
//   then wave0 s_wait_tensorcnt(0) and the block barrier publishes the tiles.
// ---------------------------------------------------------------------------
#define KV_TILE        32
#define K_ROWSTRIDE_H  136   // 128 f16 + 8 f16 TDM pad (16B) -> 272B rows
#define V_ROWSTRIDE_H  40    //  32 f16 + 8 f16 TDM pad (16B) ->  80B rows

__global__ __launch_bounds__(128)
void flash_attn_kernel(const _Float16* __restrict__ Q,
                       const _Float16* __restrict__ K,
                       const _Float16* __restrict__ Vt,   // [B, DV, S] f16
                       float* __restrict__ Out,
                       float scale) {
    __shared__ __align__(16) _Float16 ldsK[2][KV_TILE * K_ROWSTRIDE_H]; // 2 x 8.5KB
    __shared__ __align__(16) _Float16 ldsV[2][DV_ * V_ROWSTRIDE_H];     // 2 x 10KB
    __shared__ __align__(16) _Float16 ldsP[4][16 * 32];                 // per-wave P

    const int lane = threadIdx.x & 31;
    const int wave = threadIdx.x >> 5;
    const int gw   = blockIdx.x * 4 + wave;
    const int q0   = gw * 16;                 // global query row base
    const int b    = q0 / S_;
    const int half = lane >> 4;
    const int ln   = lane & 15;

    const size_t kbase = (size_t)b * S_ * DK_;
    const size_t vbase = (size_t)b * DV_ * S_;

    // Q A-fragments for full DK=128 (4 chunks of K=32), held in registers
    AFrag aq[4];
#pragma unroll
    for (int kc = 0; kc < 4; ++kc) {
        const _Float16* s0 = Q + (size_t)(q0 + ln) * DK_ + kc * 32 + half * 8;
        aq[kc].h[0] = *(const v8h*)s0;
        aq[kc].h[1] = *(const v8h*)(s0 + 16);
    }

    v8f acc[8];
#pragma unroll
    for (int i = 0; i < 8; ++i) acc[i] = vzero8f();
    float m_i[8], l_i[8];
#pragma unroll
    for (int j = 0; j < 8; ++j) { m_i[j] = -1e30f; l_i[j] = 0.0f; }

    // Prologue: stage tile 0 into buffer 0
    if (wave == 0) {
        tdm_load_2d_f16(lds_off_of(&ldsK[0][0]), K + kbase,
                        DK_, S_, DK_, KV_TILE, DK_, /*pad 256B*/5, /*+16B*/3, 1);
        tdm_load_2d_f16(lds_off_of(&ldsV[0][0]), Vt + vbase,
                        S_, DV_, KV_TILE, DV_, S_, /*pad 64B*/3, /*+16B*/3, 1);
        __builtin_amdgcn_s_wait_tensorcnt(0);
    }
    __syncthreads();

    const int NIT = S_ / KV_TILE;             // 64 iterations
    for (int it = 0; it < NIT; ++it) {
        const int cur = it & 1;
        const int kv0 = it * KV_TILE;

        // Kick off DMA for the next tile into the alternate buffers
        if (wave == 0 && it + 1 < NIT) {
            const int kv1 = kv0 + KV_TILE;
            tdm_load_2d_f16(lds_off_of(&ldsK[cur ^ 1][0]),
                            K + kbase + (size_t)kv1 * DK_,
                            DK_, S_, DK_, KV_TILE, DK_, 5, 3, 1);
            tdm_load_2d_f16(lds_off_of(&ldsV[cur ^ 1][0]),
                            Vt + vbase + kv1,
                            S_, DV_, KV_TILE, DV_, S_, 3, 3, 1);
        }

        // ---- scores: two 16x16 tiles over this 32-key strip (K from LDS) ----
        v8f st[2];
#pragma unroll
        for (int t = 0; t < 2; ++t) {
            v8f s = vzero8f();
#pragma unroll
            for (int kc = 0; kc < 4; ++kc) {
                const _Float16* bp =
                    &ldsK[cur][(t * 16 + ln) * K_ROWSTRIDE_H + kc * 32 + half * 8];
                AFrag bf;
                bf.h[0] = *(const v8h*)bp;
                bf.h[1] = *(const v8h*)(bp + 16);
                s = __builtin_amdgcn_wmma_f32_16x16x32_f16(
                    false, aq[kc].v, false, bf.v, (short)0, s, false, false);
            }
            st[t] = s;
        }

        // ---- online softmax (row = C-VGPR index j within each 16-lane half) ----
        float alpha[8];
#pragma unroll
        for (int j = 0; j < 8; ++j) {
            st[0][j] *= scale;
            st[1][j] *= scale;
            float mx = fmaxf(st[0][j], st[1][j]);
            mx = fmaxf(mx, __shfl_xor(mx, 1, 32));
            mx = fmaxf(mx, __shfl_xor(mx, 2, 32));
            mx = fmaxf(mx, __shfl_xor(mx, 4, 32));
            mx = fmaxf(mx, __shfl_xor(mx, 8, 32));
            float mnew = fmaxf(m_i[j], mx);
            alpha[j] = __expf(m_i[j] - mnew);
            float p0 = __expf(st[0][j] - mnew);
            float p1 = __expf(st[1][j] - mnew);
            st[0][j] = p0;
            st[1][j] = p1;
            float rs = p0 + p1;
            rs += __shfl_xor(rs, 1, 32);
            rs += __shfl_xor(rs, 2, 32);
            rs += __shfl_xor(rs, 4, 32);
            rs += __shfl_xor(rs, 8, 32);
            l_i[j] = l_i[j] * alpha[j] + rs;
            m_i[j] = mnew;
        }
#pragma unroll
        for (int nt = 0; nt < 8; ++nt)
#pragma unroll
            for (int j = 0; j < 8; ++j) acc[nt][j] *= alpha[j];

        // ---- P: C-layout -> LDS -> A-layout fragment ----
        _Float16* pw = ldsP[wave];
#pragma unroll
        for (int j = 0; j < 8; ++j) {
            int m = j + half * 8;
            pw[m * 32 + ln]      = (_Float16)st[0][j];
            pw[m * 32 + 16 + ln] = (_Float16)st[1][j];
        }
        __syncthreads();
        AFrag ap;
        ap.h[0] = *(const v8h*)&pw[ln * 32 + half * 8];
        ap.h[1] = *(const v8h*)&pw[ln * 32 + 16 + half * 8];
        __syncthreads();

        // ---- acc += P (16x32) x V (32x16 per N-tile), V from LDS ----
#pragma unroll
        for (int nt = 0; nt < 8; ++nt) {
            const _Float16* vp =
                &ldsV[cur][(nt * 16 + ln) * V_ROWSTRIDE_H + half * 8];
            AFrag bf;
            bf.h[0] = *(const v8h*)vp;
            bf.h[1] = *(const v8h*)(vp + 16);
            acc[nt] = __builtin_amdgcn_wmma_f32_16x16x32_f16(
                false, ap.v, false, bf.v, (short)0, acc[nt], false, false);
        }

        // Publish next tile: only wave0 has outstanding TENSORcnt
        if (wave == 0 && it + 1 < NIT) __builtin_amdgcn_s_wait_tensorcnt(0);
        __syncthreads();
    }

    // ---- epilogue: normalize by row sums, write fp32 [B, S, DV] ----
#pragma unroll
    for (int j = 0; j < 8; ++j) {
        float inv = 1.0f / l_i[j];
        int m = j + half * 8;
        size_t orow = (size_t)(q0 + m) * DV_;
#pragma unroll
        for (int nt = 0; nt < 8; ++nt)
            Out[orow + nt * 16 + ln] = acc[nt][j] * inv;
    }
}

// ---------------------------------------------------------------------------
extern "C" void kernel_launch(void* const* d_in, const int* in_sizes, int n_in,
                              void* d_out, int out_size, void* d_ws, size_t ws_size,
                              hipStream_t stream) {
    const float* x_q = (const float*)d_in[0];
    const float* x_k = (const float*)d_in[1];
    const float* x_v = (const float*)d_in[2];
    const float* Wq  = (const float*)d_in[3];
    const float* bq  = (const float*)d_in[4];
    const float* Wk  = (const float*)d_in[5];
    const float* bk  = (const float*)d_in[6];
    const float* Wv  = (const float*)d_in[7];
    const float* bv  = (const float*)d_in[8];
    float* out = (float*)d_out;

    // Workspace carve-out (all 256B-aligned): 3 f16 transposed weights,
    // q16/k16 row-major f16, V transposed f16. Total ~12.2 MB.
    char* ws = (char*)d_ws;
    _Float16* Wtq = (_Float16*)(ws);
    _Float16* Wtk = (_Float16*)(ws + 65536);
    _Float16* Wtv = (_Float16*)(ws + 131072);
    _Float16* q16 = (_Float16*)(ws + 196608);
    _Float16* k16 = (_Float16*)(ws + 196608 + (size_t)4194304);
    _Float16* vT  = (_Float16*)(ws + 196608 + (size_t)2 * 4194304);

    const int wtBlocks = (D_ * DK_ + 255) / 256;
    wt_convert_kernel<<<wtBlocks, 256, 0, stream>>>(Wq, Wtq);
    wt_convert_kernel<<<wtBlocks, 256, 0, stream>>>(Wk, Wtk);
    wt_convert_kernel<<<wtBlocks, 256, 0, stream>>>(Wv, Wtv);

    const int projBlocks = (B_ * S_) / (16 * 8);   // 128 blocks x 8 waves x 16 rows
    proj16_kernel<<<projBlocks, 256, 0, stream>>>(x_q, Wtq, bq, q16, 0);
    proj16_kernel<<<projBlocks, 256, 0, stream>>>(x_k, Wtk, bk, k16, 0);
    proj16_kernel<<<projBlocks, 256, 0, stream>>>(x_v, Wtv, bv, vT, 1);

    const float scale = 0.08838834764831843f;      // 1/sqrt(128)
    const int faBlocks = (B_ * S_) / 64;           // 256 blocks x 4 waves x 16 rows
    flash_attn_kernel<<<faBlocks, 128, 0, stream>>>(q16, k16, vT, out, scale);
}
